// SpatiaTemporallNeighbourhoodAttentionBlock_24541443130185
// MI455X (gfx1250) — matrix operand order repaired and checked by reading, not compiled, not run on hardware
//
#include <hip/hip_runtime.h>
#include <hip/hip_bf16.h>

// ---------------------------------------------------------------------------
// 3D Neighborhood Attention block for MI455X (gfx1250, wave32, WMMA).
//   B=1, C=128, T=12, H=40, W=40, kernel 3x3x3 (27 neighbors, border-clamped)
// Pipeline: RMS-norm -> QKV GEMM (WMMA f16) -> NATTEN softmax-attn (VALU f16)
//           -> proj GEMM (WMMA f16) fused with bias + residual + transpose.
// ---------------------------------------------------------------------------

typedef __attribute__((ext_vector_type(16))) _Float16 v16h;
typedef __attribute__((ext_vector_type(8)))  _Float16 v8h;
typedef __attribute__((ext_vector_type(4)))  _Float16 v4h;
typedef __attribute__((ext_vector_type(8)))  float    v8f;

#define LTOK   19200   // T*H*W
#define CCH    128
#define TT     12
#define HH     40
#define WW     40
#define INV_SQRT_C 0.08838834764831845f  // 128^-0.5
#define SQRT_C     11.313708498984761f   // 128^0.5

// ---- WMMA fragment loaders (contiguous k order, shared by A and B) ------------------
// A 16x32 f16: lanes 0-15 row m=lane, K[0..15];
//              lanes 16-31 row m=lane-16, K[16..31].
__device__ __forceinline__ v16h load_a_frag(const _Float16* __restrict__ row_ptr,
                                            int kofs /* 0 or 16 per lane half */) {
  return *(const v16h*)(row_ptr + kofs);
}
// B 32x16 f16 (contiguous K-blocks, same k order as A):
// lanes 0-15 col n=lane, halves K[0..15]; lanes 16-31 col n=lane-16, K[16..31].
__device__ __forceinline__ v16h load_b_frag(const _Float16* __restrict__ col_ptr,
                                            int kofs /* 0 or 16 per lane half */) {
  return *(const v16h*)(col_ptr + kofs);
}

// ---- Kernel 1: convert f32 weights to f16 ----------------------------------
__global__ void cvt_weights(const float* __restrict__ qkv_w,
                            const float* __restrict__ proj_w,
                            _Float16* __restrict__ qw16,
                            _Float16* __restrict__ pw16) {
  int i = blockIdx.x * blockDim.x + threadIdx.x;
  if (i < 384 * CCH)  qw16[i] = (_Float16)qkv_w[i];
  if (i < CCH * CCH)  pw16[i] = (_Float16)proj_w[i];
}

// ---- Kernel 2: RMS-style channel norm, channels-first f32 -> (L,C) f16 -----
__global__ void rms_norm(const float* __restrict__ x,
                         const float* __restrict__ gamma,
                         _Float16* __restrict__ xn) {
  int pos = blockIdx.x * blockDim.x + threadIdx.x;
  if (pos >= LTOK) return;
  float ss = 0.f;
#pragma unroll 4
  for (int c = 0; c < CCH; ++c) {
    float t = x[c * LTOK + pos];
    ss += t * t;
  }
  // xn = x / max(||x||, 1e-12) * sqrt(C) * gamma
  float inv = SQRT_C * rsqrtf(fmaxf(ss, 1e-24f));
#pragma unroll 4
  for (int c = 0; c < CCH; ++c) {
    xn[pos * CCH + c] = (_Float16)(x[c * LTOK + pos] * inv * gamma[c]);
  }
}

// ---- Kernel 3: QKV GEMM, (L,128) x (128,384)^T, one wave = 16x64 tile ------
__global__ void qkv_gemm(const _Float16* __restrict__ xn,
                         const _Float16* __restrict__ w16,
                         const float* __restrict__ qkv_b,
                         _Float16* __restrict__ q,
                         _Float16* __restrict__ k,
                         _Float16* __restrict__ v) {
  const int NGRP = 6;                       // 384 / 64
  int wave = (blockIdx.x * blockDim.x + threadIdx.x) >> 5;
  int lane = threadIdx.x & 31;
  int mtile = wave / NGRP;
  int ngrp  = wave % NGRP;
  if (mtile >= LTOK / 16) return;

  int arow   = mtile * 16 + (lane & 15);
  int khalfA = (lane & 16) ? 16 : 0;
  int khalfB = (lane & 16) ? 16 : 0;
  const _Float16* aptr = xn + arow * CCH;

  v8f acc[4] = {v8f{}, v8f{}, v8f{}, v8f{}};
#pragma unroll
  for (int kk = 0; kk < 4; ++kk) {          // K = 128 = 4 x 32
    v16h a = load_a_frag(aptr + kk * 32, khalfA);
#pragma unroll
    for (int s = 0; s < 4; ++s) {           // 4 N-subtiles reuse A
      int wrow = ngrp * 64 + s * 16 + (lane & 15);
      v16h b = load_b_frag(w16 + wrow * CCH + kk * 32, khalfB);
      acc[s] = __builtin_amdgcn_wmma_f32_16x16x32_f16(
          false, a, false, b, (short)0, acc[s], false, false);
    }
  }

  // epilogue: bias, q-scale, split into q/k/v (f16, row-major (L,C))
  _Float16* dst = (ngrp < 2) ? q : (ngrp < 4) ? k : v;
  int rbase = mtile * 16 + ((lane & 16) ? 8 : 0);
#pragma unroll
  for (int s = 0; s < 4; ++s) {
    int col   = ngrp * 64 + s * 16 + (lane & 15);
    float bi  = qkv_b[col];
    float sc  = (col < CCH) ? INV_SQRT_C : 1.f;   // q = q * C^-0.5
    int   lc  = col & (CCH - 1);
#pragma unroll
    for (int r = 0; r < 8; ++r) {
      dst[(rbase + r) * CCH + lc] = (_Float16)((acc[s][r] + bi) * sc);
    }
  }
}

// ---- Kernel 4: 3x3x3 neighborhood attention, one wave per query ------------
__global__ void natten3d(const _Float16* __restrict__ q,
                         const _Float16* __restrict__ k,
                         const _Float16* __restrict__ v,
                         const float* __restrict__ rpb,  // (5,5,5)
                         _Float16* __restrict__ ao) {
  int wv   = threadIdx.x >> 5;
  int lane = threadIdx.x & 31;
  int pos  = blockIdx.x * 8 + wv;
  if (pos >= LTOK) return;

  int t   = pos / (HH * WW);
  int rem = pos - t * (HH * WW);
  int h   = rem / WW;
  int w   = rem - h * WW;
  int st  = min(max(t - 1, 0), TT - 3);
  int sh  = min(max(h - 1, 0), HH - 3);
  int sw  = min(max(w - 1, 0), WW - 3);

  // phase 1: lane j owns neighbor j (j < 27): 128-dim q.k dot + rpb bias
  float logit = -3.0e38f;
  int   nbr   = 0;
  if (lane < 27) {
    int dt = lane / 9, dh = (lane / 3) % 3, dw = lane % 3;
    int nt = st + dt, nh = sh + dh, nw = sw + dw;
    nbr = (nt * HH + nh) * WW + nw;
    const _Float16* qp = q + pos * CCH;
    const _Float16* kp = k + nbr * CCH;
    float dot = 0.f;
#pragma unroll
    for (int c = 0; c < CCH; c += 8) {
      v8h qa = *(const v8h*)(qp + c);
      v8h ka = *(const v8h*)(kp + c);
#pragma unroll
      for (int i = 0; i < 8; ++i) dot += (float)qa[i] * (float)ka[i];
    }
    logit = dot + rpb[((nt - t + 2) * 5 + (nh - h + 2)) * 5 + (nw - w + 2)];
  }

  // softmax over the 27 lanes via wave32 shuffle reductions
  float m = logit;
#pragma unroll
  for (int off = 16; off; off >>= 1) m = fmaxf(m, __shfl_xor(m, off, 32));
  float p = (lane < 27) ? __expf(logit - m) : 0.f;
  float s = p;
#pragma unroll
  for (int off = 16; off; off >>= 1) s += __shfl_xor(s, off, 32);
  float a = p / s;

  // phase 2: lane owns 4 channels, broadcast (a_j, nbr_j) via readlane shuffle
  int c0 = lane * 4;
  float acc0 = 0.f, acc1 = 0.f, acc2 = 0.f, acc3 = 0.f;
#pragma unroll 9
  for (int j = 0; j < 27; ++j) {
    float aj = __shfl(a, j, 32);
    int   nb = __shfl(nbr, j, 32);
    v4h vv = *(const v4h*)(v + nb * CCH + c0);
    acc0 += aj * (float)vv[0];
    acc1 += aj * (float)vv[1];
    acc2 += aj * (float)vv[2];
    acc3 += aj * (float)vv[3];
  }
  v4h o;
  o[0] = (_Float16)acc0; o[1] = (_Float16)acc1;
  o[2] = (_Float16)acc2; o[3] = (_Float16)acc3;
  *(v4h*)(ao + pos * CCH + c0) = o;
}

// ---- Kernel 5: proj GEMM + bias + residual, writes channels-first f32 ------
__global__ void proj_gemm(const _Float16* __restrict__ ao,
                          const _Float16* __restrict__ w16,
                          const float* __restrict__ proj_b,
                          const float* __restrict__ x,
                          float* __restrict__ out) {
  const int NGRP = 2;                       // 128 / 64
  int wave = (blockIdx.x * blockDim.x + threadIdx.x) >> 5;
  int lane = threadIdx.x & 31;
  int mtile = wave / NGRP;
  int ngrp  = wave % NGRP;
  if (mtile >= LTOK / 16) return;

  int arow   = mtile * 16 + (lane & 15);
  int khalfA = (lane & 16) ? 16 : 0;
  int khalfB = (lane & 16) ? 16 : 0;
  const _Float16* aptr = ao + arow * CCH;

  v8f acc[4] = {v8f{}, v8f{}, v8f{}, v8f{}};
#pragma unroll
  for (int kk = 0; kk < 4; ++kk) {
    v16h a = load_a_frag(aptr + kk * 32, khalfA);
#pragma unroll
    for (int s = 0; s < 4; ++s) {
      int wrow = ngrp * 64 + s * 16 + (lane & 15);
      v16h b = load_b_frag(w16 + wrow * CCH + kk * 32, khalfB);
      acc[s] = __builtin_amdgcn_wmma_f32_16x16x32_f16(
          false, a, false, b, (short)0, acc[s], false, false);
    }
  }

  int rbase = mtile * 16 + ((lane & 16) ? 8 : 0);
#pragma unroll
  for (int s = 0; s < 4; ++s) {
    int col  = ngrp * 64 + s * 16 + (lane & 15);
    float bi = proj_b[col];
#pragma unroll
    for (int r = 0; r < 8; ++r) {
      int row = rbase + r;                  // token index
      out[col * LTOK + row] = acc[s][r] + bi + x[col * LTOK + row];
    }
  }
}

// ---------------------------------------------------------------------------
extern "C" void kernel_launch(void* const* d_in, const int* in_sizes, int n_in,
                              void* d_out, int out_size, void* d_ws, size_t ws_size,
                              hipStream_t stream) {
  const float* x      = (const float*)d_in[0];  // (1,128,12,40,40)
  const float* gamma  = (const float*)d_in[1];  // (128,)
  const float* qkv_w  = (const float*)d_in[2];  // (384,128)
  const float* qkv_b  = (const float*)d_in[3];  // (384,)
  const float* rpb    = (const float*)d_in[4];  // (1,5,5,5)
  const float* proj_w = (const float*)d_in[5];  // (128,128)
  const float* proj_b = (const float*)d_in[6];  // (128,)
  float* out = (float*)d_out;

  // workspace carve-up (f16 intermediates)
  char* ws = (char*)d_ws;
  const size_t LC = (size_t)LTOK * CCH * sizeof(_Float16);  // 4.9 MB each
  _Float16* xn  = (_Float16*)(ws);
  _Float16* qb  = (_Float16*)(ws + 1 * LC);
  _Float16* kb  = (_Float16*)(ws + 2 * LC);
  _Float16* vb  = (_Float16*)(ws + 3 * LC);
  _Float16* ab  = (_Float16*)(ws + 4 * LC);
  _Float16* qw16 = (_Float16*)(ws + 5 * LC);
  _Float16* pw16 = (_Float16*)(ws + 5 * LC + 384 * CCH * sizeof(_Float16));

  cvt_weights<<<192, 256, 0, stream>>>(qkv_w, proj_w, qw16, pw16);
  rms_norm<<<(LTOK + 255) / 256, 256, 0, stream>>>(x, gamma, xn);
  // 1200 M-tiles x 6 N-groups = 7200 waves, 8 waves/block
  qkv_gemm<<<900, 256, 0, stream>>>(xn, qw16, qkv_b, qb, kb, vb);
  // one wave per token
  natten3d<<<LTOK / 8, 256, 0, stream>>>(qb, kb, vb, rpb, ab);
  // 1200 M-tiles x 2 N-groups = 2400 waves
  proj_gemm<<<300, 256, 0, stream>>>(ab, pw16, proj_b, x, out);
}
